// MRN4GRUCell_7189775254025
// MI455X (gfx1250) — matrix operand order repaired and checked by reading, not compiled, not run on hardware
//
#include <hip/hip_runtime.h>
#include <hip/hip_bf16.h>

typedef __attribute__((ext_vector_type(16))) __bf16 v16bf;
typedef __attribute__((ext_vector_type(8)))  float  v8f;
typedef __attribute__((ext_vector_type(4)))  unsigned int u32x4;
typedef __attribute__((ext_vector_type(8)))  unsigned int u32x8;

#define BB 4096
#define II 512
#define HH 512
#define TT 8
#define G3 1536   // 3*H
#define K1 4096   // T*H

union BF16Frag { uint4 u[2]; v16bf v; };

__device__ __forceinline__ unsigned int f2bf(float f) {
  unsigned int u = __float_as_uint(f);
  u += 0x7FFFu + ((u >> 16) & 1u);   // round-to-nearest-even
  return u >> 16;
}

__device__ __forceinline__ v8f zerov8() { return (v8f)(0.0f); }

// A fragment (16x32 bf16, M x K): lane m holds K = base..base+15,
// base = k0 + (hi?16:0). Two 16B LDS loads.
__device__ __forceinline__ v16bf ldfragA(const unsigned short* rowp, int base) {
  BF16Frag f;
  f.u[0] = *reinterpret_cast<const uint4*>(rowp + base);
  f.u[1] = *reinterpret_cast<const uint4*>(rowp + base + 8);
  return f.v;
}

// B fragment (32x16 bf16, K x N): lane n holds 16 contiguous K of weight row n.
__device__ __forceinline__ v16bf ldfragB(const unsigned short* p) {
  BF16Frag f;
  f.u[0] = *reinterpret_cast<const uint4*>(p);
  f.u[1] = *reinterpret_cast<const uint4*>(p + 8);
  return f.v;
}

__device__ __forceinline__ v8f mma(v16bf a, v16bf b, v8f c) {
  return __builtin_amdgcn_wmma_f32_16x16x32_bf16(false, a, false, b, (short)0, c, false, false);
}

__device__ __forceinline__ float sigmoidf_(float v) { return 1.0f / (1.0f + __expf(-v)); }

// TDM: DMA one 2D bf16 tile (tile_dim0=128 elems x tile_dim1=32 rows) from a
// (4096 x 4096)-element bf16 tensor (row stride 4096 elems) into LDS.
// D# group0: count=1 | lds_addr | global_addr[56:0] | type=2
// D# group1: data_size=2B, tensor_dim0/1=4096, tile_dim0=128, tile_dim1=32,
//            tensor_dim0_stride=4096 (bit-exact packing per CDNA5 ISA 8.3/8.4).
__device__ __forceinline__ void tdm_load_tile_32x128(unsigned ldsOff, const unsigned short* gptr) {
  const unsigned long long ga = (unsigned long long)(uintptr_t)gptr;
  u32x4 g0;
  g0.x = 1u;                                          // count=1, user mode
  g0.y = ldsOff;                                      // lds_addr (bytes)
  g0.z = (unsigned int)ga;                            // global_addr[31:0]
  g0.w = ((unsigned int)(ga >> 32) & 0x01FFFFFFu)     // global_addr[56:32]
         | (2u << 30);                                // type=2 (image)
  u32x8 g1;
  g1.s0 = (1u << 16);                                 // workgroup_mask=0, data_size=1 (2 bytes)
  g1.s1 = (4096u & 0xFFFFu) << 16;                    // tensor_dim0[15:0] @ bits 63:48
  g1.s2 = (4096u >> 16) | ((4096u & 0xFFFFu) << 16);  // tensor_dim0[31:16] | tensor_dim1[15:0]
  g1.s3 = (4096u >> 16) | (128u << 16);               // tensor_dim1[31:16] | tile_dim0=128
  g1.s4 = 32u;                                        // tile_dim1=32, tile_dim2=0
  g1.s5 = 4096u;                                      // tensor_dim0_stride[31:0]
  g1.s6 = 0u;                                         // stride0[47:32]=0 | stride1[15:0]=0
  g1.s7 = 0u;                                         // stride1[47:16]=0
  asm volatile("tensor_load_to_lds %0, %1" :: "s"(g0), "s"(g1) : "memory");
}

// ---------------- utility kernels ----------------

__global__ void k_zero_counts(int* counts) {
  if (threadIdx.x < TT) counts[threadIdx.x] = 0;
}

__global__ void k_bucket(const int* __restrict__ type, int* counts, int* lists) {
  const int i = blockIdx.x * blockDim.x + threadIdx.x;
  if (i < BB) {
    const int t = type[i];
    const int p = atomicAdd(&counts[t], 1);
    lists[t * BB + p] = i;
  }
}

__global__ void k_copy_f4(const float4* __restrict__ src, float4* __restrict__ dst, int n4) {
  const int i = blockIdx.x * blockDim.x + threadIdx.x;
  if (i < n4) dst[i] = src[i];
}

// f32 -> bf16, 8 elements per thread
__global__ void k_cvt_bf16(const float* __restrict__ src, unsigned short* __restrict__ dst, int n) {
  const int i = blockIdx.x * blockDim.x + threadIdx.x;
  if (i * 8 < n) {
    float4 a = reinterpret_cast<const float4*>(src)[i * 2];
    float4 b = reinterpret_cast<const float4*>(src)[i * 2 + 1];
    uint4 o;
    o.x = f2bf(a.x) | (f2bf(a.y) << 16);
    o.y = f2bf(a.z) | (f2bf(a.w) << 16);
    o.z = f2bf(b.x) | (f2bf(b.y) << 16);
    o.w = f2bf(b.z) | (f2bf(b.w) << 16);
    reinterpret_cast<uint4*>(dst)[i] = o;
  }
}

// ---------------- stage A: per-type GRU cells on bucketed rows ----------------
// grid: (hTile 0..7, rowTile 0..255, type 0..7), block 128 = 4 waves, M=16 N=64.

__global__ __launch_bounds__(128) void k_stageA(
    const float* __restrict__ x, const float* __restrict__ regs,
    const unsigned short* __restrict__ Wih, const unsigned short* __restrict__ Whh,
    const float* __restrict__ b_ih, const float* __restrict__ b_hh,
    const int* __restrict__ counts, const int* __restrict__ lists,
    float* __restrict__ out_regs)
{
  const int hTile = blockIdx.x, rowTile = blockIdx.y, t = blockIdx.z;
  const int cnt = counts[t];
  if (rowTile * 16 >= cnt) return;
  const int rem = cnt - rowTile * 16;
  const int nv = rem < 16 ? rem : 16;

  __shared__ __align__(16) unsigned short Ax[16 * II];
  __shared__ __align__(16) unsigned short Ah[16 * HH];
  __shared__ int ridx[16];

  const int tid = threadIdx.x;
  if (tid < 16) {
    const int mm = tid < nv ? tid : nv - 1;
    ridx[tid] = lists[t * BB + rowTile * 16 + mm];
  }
  __syncthreads();

  for (int e = tid; e < 16 * (II / 4); e += 128) {
    const int m = e >> 7, k4 = e & 127;
    const int r = ridx[m];
    const float4 xv = reinterpret_cast<const float4*>(x + (size_t)r * II)[k4];
    const float4 hv = reinterpret_cast<const float4*>(regs + ((size_t)r * TT + t) * HH)[k4];
    uint2 px, ph;
    px.x = f2bf(xv.x) | (f2bf(xv.y) << 16);
    px.y = f2bf(xv.z) | (f2bf(xv.w) << 16);
    ph.x = f2bf(hv.x) | (f2bf(hv.y) << 16);
    ph.y = f2bf(hv.z) | (f2bf(hv.w) << 16);
    *reinterpret_cast<uint2*>(Ax + m * II + k4 * 4) = px;
    *reinterpret_cast<uint2*>(Ah + m * HH + k4 * 4) = ph;
  }
  __syncthreads();

  const int wave = tid >> 5, lane = tid & 31;
  const int n = lane & 15, hi = lane >> 4;
  const int col = hTile * 64 + wave * 16 + n;

  v8f gir = zerov8(), giz = zerov8(), gin = zerov8();
  v8f ghr = zerov8(), ghz = zerov8(), ghn = zerov8();

  const unsigned short* Wr = Wih + ((size_t)t * G3 + col) * II;
  const unsigned short* Wz = Wih + ((size_t)t * G3 + HH + col) * II;
  const unsigned short* Wn = Wih + ((size_t)t * G3 + 2 * HH + col) * II;
  const unsigned short* Ur = Whh + ((size_t)t * G3 + col) * HH;
  const unsigned short* Uz = Whh + ((size_t)t * G3 + HH + col) * HH;
  const unsigned short* Un = Whh + ((size_t)t * G3 + 2 * HH + col) * HH;

  for (int k0 = 0; k0 < II; k0 += 32) {
    const int ab = k0 + hi * 16, bb = k0 + hi * 16;
    const v16bf ax = ldfragA(Ax + n * II, ab);
    const v16bf ah = ldfragA(Ah + n * HH, ab);
    const v16bf br = ldfragB(Wr + bb);
    const v16bf bz = ldfragB(Wz + bb);
    const v16bf bn = ldfragB(Wn + bb);
    const v16bf ur = ldfragB(Ur + bb);
    const v16bf uz = ldfragB(Uz + bb);
    const v16bf un = ldfragB(Un + bb);
    gir = mma(ax, br, gir);
    giz = mma(ax, bz, giz);
    gin = mma(ax, bn, gin);
    ghr = mma(ah, ur, ghr);
    ghz = mma(ah, uz, ghz);
    ghn = mma(ah, un, ghn);
  }

  const float bir = b_ih[t * G3 + col];
  const float biz = b_ih[t * G3 + HH + col];
  const float bin_ = b_ih[t * G3 + 2 * HH + col];
  const float bhr = b_hh[t * G3 + col];
  const float bhz = b_hh[t * G3 + HH + col];
  const float bhn = b_hh[t * G3 + 2 * HH + col];

#pragma unroll
  for (int g = 0; g < 8; ++g) {
    const int m = g + hi * 8;
    if (m < nv) {
      const float r = sigmoidf_(gir[g] + bir + ghr[g] + bhr);
      const float z = sigmoidf_(giz[g] + biz + ghz[g] + bhz);
      const float nn = tanhf(gin[g] + bin_ + r * (ghn[g] + bhn));
      const size_t off = ((size_t)ridx[m] * TT + t) * HH + col;
      out_regs[off] = (1.0f - z) * nn + z * regs[off];
    }
  }
}

// ---------------- stage 2: core GRU, M=32 x N=64 tiles, TDM-staged A ----------------
// grid: (hTile 0..7, rowTile 0..127), block 128 = 4 waves.
// A chunks (32x128 bf16) are DMA'd into double-buffered LDS by the Tensor Data
// Mover (wave 0 issues tensor_load_to_lds; TENSORcnt + barrier synchronize),
// overlapping with B-fragment loads + 6 wmmas per 32-K step.

__global__ __launch_bounds__(128) void k_stage2(
    const unsigned short* __restrict__ Xbf,   // (B, 4096) bf16 new_registers
    const float* __restrict__ c,              // (B, 512)
    const unsigned short* __restrict__ cWih,  // (1536, 4096) bf16
    const unsigned short* __restrict__ cWhh,  // (1536, 512) bf16
    const float* __restrict__ cb_ih, const float* __restrict__ cb_hh,
    float* __restrict__ core)                 // (B, 512)
{
  const int hTile = blockIdx.x, rowTile = blockIdx.y;
  const int row0 = rowTile * 32;
  __shared__ __align__(128) unsigned short As[2][32 * 128];  // double-buffered A chunks

  const int tid = threadIdx.x;
  const int wave = tid >> 5, lane = tid & 31;
  const int n = lane & 15, hi = lane >> 4;
  const int col = hTile * 64 + wave * 16 + n;

  v8f gir0 = zerov8(), giz0 = zerov8(), gin0 = zerov8();
  v8f gir1 = zerov8(), giz1 = zerov8(), gin1 = zerov8();

  const unsigned short* Wr = cWih + (size_t)col * K1;
  const unsigned short* Wz = cWih + ((size_t)(HH + col)) * K1;
  const unsigned short* Wn = cWih + ((size_t)(2 * HH + col)) * K1;

  const unsigned ldsA0 = (unsigned)(uintptr_t)&As[0][0];
  const unsigned ldsA1 = (unsigned)(uintptr_t)&As[1][0];
  const unsigned short* Xrow = Xbf + (size_t)row0 * K1;

  // prologue: DMA chunk 0 into buffer 0
  if (wave == 0) {
    tdm_load_tile_32x128(ldsA0, Xrow);
    __builtin_amdgcn_s_wait_tensorcnt(0);
  }
  __syncthreads();

  for (int cchunk = 0; cchunk < 32; ++cchunk) {
    const int kc = cchunk * 128;
    if (cchunk + 1 < 32) {
      if (wave == 0) {
        // DMA next chunk into the other buffer while everyone computes
        tdm_load_tile_32x128((cchunk & 1) ? ldsA0 : ldsA1, Xrow + kc + 128);
      }
      // prefetch next chunk's weight rows -> global_prefetch_b8
      __builtin_prefetch(Wr + kc + 128, 0, 1);
      __builtin_prefetch(Wz + kc + 128, 0, 1);
      __builtin_prefetch(Wn + kc + 128, 0, 1);
    }
    const unsigned short* Ab = As[cchunk & 1];
#pragma unroll
    for (int k0 = 0; k0 < 128; k0 += 32) {
      const int ab = k0 + hi * 16;
      const int bb = kc + k0 + hi * 16;
      const v16bf a0 = ldfragA(Ab + n * 128, ab);
      const v16bf a1 = ldfragA(Ab + (n + 16) * 128, ab);
      const v16bf br = ldfragB(Wr + bb);
      const v16bf bz = ldfragB(Wz + bb);
      const v16bf bn = ldfragB(Wn + bb);
      gir0 = mma(a0, br, gir0);
      gir1 = mma(a1, br, gir1);
      giz0 = mma(a0, bz, giz0);
      giz1 = mma(a1, bz, giz1);
      gin0 = mma(a0, bn, gin0);
      gin1 = mma(a1, bn, gin1);
    }
    if (wave == 0) __builtin_amdgcn_s_wait_tensorcnt(0);
    __syncthreads();
  }

  // ---- gh over K = 512 (convert c on the fly into As[0]) ----
  v8f ghr0 = zerov8(), ghz0 = zerov8(), ghn0 = zerov8();
  v8f ghr1 = zerov8(), ghz1 = zerov8(), ghn1 = zerov8();
  const unsigned short* Vr = cWhh + (size_t)col * HH;
  const unsigned short* Vz = cWhh + ((size_t)(HH + col)) * HH;
  const unsigned short* Vn = cWhh + ((size_t)(2 * HH + col)) * HH;

  for (int kc = 0; kc < HH; kc += 128) {
#pragma unroll
    for (int it = 0; it < 8; ++it) {   // 32*128 f32 = 1024 float4, 8 per thread
      const int e = tid + it * 128;
      const int m = e >> 5, k4 = e & 31;
      const float4 v = reinterpret_cast<const float4*>(c + ((size_t)(row0 + m)) * HH + kc)[k4];
      uint2 p;
      p.x = f2bf(v.x) | (f2bf(v.y) << 16);
      p.y = f2bf(v.z) | (f2bf(v.w) << 16);
      *reinterpret_cast<uint2*>(As[0] + m * 128 + k4 * 4) = p;
    }
    __syncthreads();
#pragma unroll
    for (int k0 = 0; k0 < 128; k0 += 32) {
      const int ab = k0 + hi * 16;
      const int bb = kc + k0 + hi * 16;
      const v16bf a0 = ldfragA(As[0] + n * 128, ab);
      const v16bf a1 = ldfragA(As[0] + (n + 16) * 128, ab);
      const v16bf br = ldfragB(Vr + bb);
      const v16bf bz = ldfragB(Vz + bb);
      const v16bf bn = ldfragB(Vn + bb);
      ghr0 = mma(a0, br, ghr0);
      ghr1 = mma(a1, br, ghr1);
      ghz0 = mma(a0, bz, ghz0);
      ghz1 = mma(a1, bz, ghz1);
      ghn0 = mma(a0, bn, ghn0);
      ghn1 = mma(a1, bn, ghn1);
    }
    __syncthreads();
  }

  const float bir = cb_ih[col], biz = cb_ih[HH + col], bin_ = cb_ih[2 * HH + col];
  const float bhr = cb_hh[col], bhz = cb_hh[HH + col], bhn = cb_hh[2 * HH + col];

#pragma unroll
  for (int g = 0; g < 8; ++g) {
    const int m = g + hi * 8;
    {
      const int row = row0 + m;
      const float r = sigmoidf_(gir0[g] + bir + ghr0[g] + bhr);
      const float z = sigmoidf_(giz0[g] + biz + ghz0[g] + bhz);
      const float nn = tanhf(gin0[g] + bin_ + r * (ghn0[g] + bhn));
      const float hp = c[(size_t)row * HH + col];
      core[(size_t)row * HH + col] = (1.0f - z) * nn + z * hp;
    }
    {
      const int row = row0 + 16 + m;
      const float r = sigmoidf_(gir1[g] + bir + ghr1[g] + bhr);
      const float z = sigmoidf_(giz1[g] + biz + ghz1[g] + bhz);
      const float nn = tanhf(gin1[g] + bin_ + r * (ghn1[g] + bhn));
      const float hp = c[(size_t)row * HH + col];
      core[(size_t)row * HH + col] = (1.0f - z) * nn + z * hp;
    }
  }
}

// ---------------- launcher ----------------

extern "C" void kernel_launch(void* const* d_in, const int* in_sizes, int n_in,
                              void* d_out, int out_size, void* d_ws, size_t ws_size,
                              hipStream_t stream) {
  (void)in_sizes; (void)n_in; (void)out_size; (void)ws_size;

  const float* x     = (const float*)d_in[0];
  const int*   type  = (const int*)d_in[1];
  const float* regs  = (const float*)d_in[2];
  const float* c     = (const float*)d_in[3];
  const float* W_ih  = (const float*)d_in[4];
  const float* W_hh  = (const float*)d_in[5];
  const float* b_ih  = (const float*)d_in[6];
  const float* b_hh  = (const float*)d_in[7];
  const float* cW_ih = (const float*)d_in[8];
  const float* cW_hh = (const float*)d_in[9];
  const float* cb_ih = (const float*)d_in[10];
  const float* cb_hh = (const float*)d_in[11];

  float* out_regs = (float*)d_out;                        // (B, T, H)
  float* out_core = out_regs + (size_t)BB * TT * HH;      // (B, H)

  char* ws = (char*)d_ws;
  int* counts = (int*)ws;
  int* lists  = (int*)(ws + 256);
  size_t off = 256 + (size_t)TT * BB * sizeof(int);
  unsigned short* Wih_bf  = (unsigned short*)(ws + off); off += (size_t)TT * G3 * II * 2;
  unsigned short* Whh_bf  = (unsigned short*)(ws + off); off += (size_t)TT * G3 * HH * 2;
  unsigned short* cWih_bf = (unsigned short*)(ws + off); off += (size_t)G3 * K1 * 2;
  unsigned short* cWhh_bf = (unsigned short*)(ws + off); off += (size_t)G3 * HH * 2;
  unsigned short* Xbf     = (unsigned short*)(ws + off);

  k_zero_counts<<<1, 32, 0, stream>>>(counts);
  k_bucket<<<BB / 256, 256, 0, stream>>>(type, counts, lists);

  {
    const int n4 = BB * TT * HH / 4;
    k_copy_f4<<<n4 / 256, 256, 0, stream>>>((const float4*)regs, (float4*)out_regs, n4);
  }

  k_cvt_bf16<<<(TT * G3 * II) / 2048, 256, 0, stream>>>(W_ih, Wih_bf, TT * G3 * II);
  k_cvt_bf16<<<(TT * G3 * HH) / 2048, 256, 0, stream>>>(W_hh, Whh_bf, TT * G3 * HH);
  k_cvt_bf16<<<(G3 * K1) / 2048, 256, 0, stream>>>(cW_ih, cWih_bf, G3 * K1);
  k_cvt_bf16<<<(G3 * HH) / 2048, 256, 0, stream>>>(cW_hh, cWhh_bf, G3 * HH);

  k_stageA<<<dim3(8, BB / 16, TT), 128, 0, stream>>>(
      x, regs, Wih_bf, Whh_bf, b_ih, b_hh, counts, lists, out_regs);

  k_cvt_bf16<<<(BB * TT * HH) / 2048, 256, 0, stream>>>(out_regs, Xbf, BB * TT * HH);

  k_stage2<<<dim3(8, BB / 32), 128, 0, stream>>>(
      Xbf, c, cWih_bf, cWhh_bf, cb_ih, cb_hh, out_core);
}
